// NeuralMemory_39694087750286
// MI455X (gfx1250) — matrix operand order repaired and checked by reading, not compiled, not run on hardware
//
#include <hip/hip_runtime.h>
#include <hip/hip_bf16.h>
#include <stdint.h>

typedef __attribute__((ext_vector_type(16))) _Float16 v16h;
typedef __attribute__((ext_vector_type(8)))  _Float16 v8h;
typedef __attribute__((ext_vector_type(8)))  float    v8f;

#define DEV __device__ __forceinline__

// ---- constants ----
#define BSZ   8
#define SEQ   4096
#define DIM   512
#define NH    8
#define DH    64
#define NCH   64           // n = SEQ/CHUNK
#define CHK   64
#define BH    64           // BSZ*NH
#define ROWS  32768        // BSZ*SEQ

// ---------------- WMMA fragment packing (CDNA5 ISA §7.12.2) ----------------
// A (16x32 f16): lanes 0-15 rows; per-lane K: {0..15} (lo half) / {16..31}
DEV int a_koff(int j, int lane) { return ((lane >> 4) << 4) + j; }
// B (32x16 f16): lanes 0-15 hold K=0..15, lanes 16-31 hold K=16..31 (col = lane&15)
DEV int b_koff(int j, int lane) { return ((lane >> 4) << 4) + j; }

// A from row-major [M x K] storage — contiguous per lane (2 x ds_load_b128)
DEV v16h pack_a(const _Float16* src, int ld, int m0, int k0, int lane) {
  v16h r; int m = m0 + (lane & 15);
#pragma unroll
  for (int j = 0; j < 16; ++j) r[j] = src[m * ld + k0 + a_koff(j, lane)];
  return r;
}
// B from COLUMN-major storage src[n][k] (i.e. B[k][n] = src[n*ld+k]) — contiguous per lane
DEV v16h pack_bt(const _Float16* src, int ld, int k0, int n0, int lane) {
  v16h r; int n = n0 + (lane & 15);
#pragma unroll
  for (int j = 0; j < 16; ++j) r[j] = src[n * ld + k0 + b_koff(j, lane)];
  return r;
}
DEV v8f wmma16(v16h a, v16h b, v8f c) {
  return __builtin_amdgcn_wmma_f32_16x16x32_f16(false, a, false, b, (short)0, c,
                                                false, false);
}
DEV float sigmoidf_(float x) { return 1.f / (1.f + __expf(-x)); }
DEV float cleanf_(float x)   { return (x - x == 0.f) ? x : 0.f; }   // nan/inf -> 0

// ---- async global -> LDS (CDNA5 ISA, GV mode) ----
// VDST = per-lane LDS byte address, VADDR = 64-bit global address; tracked by ASYNCcnt.
DEV void async_ld_b128(void* lds_dst, const void* gsrc) {
  uint32_t loff = (uint32_t)(size_t)lds_dst;
  asm volatile("global_load_async_to_lds_b128 %0, %1, off"
               :: "v"(loff), "v"(gsrc) : "memory");
}
DEV void wait_async0() { asm volatile("s_wait_asynccnt 0x0" ::: "memory"); }

// ---------------- Kernel 0a: f32 -> f16 convert (row-major) ----------------
__global__ __launch_bounds__(256) void nm_cvt_f16(const float* __restrict__ src,
                                                  _Float16* __restrict__ dst, int count) {
  int i = blockIdx.x * 256 + threadIdx.x;
  if (i < count) dst[i] = (_Float16)src[i];
}
// ---------------- Kernel 0b: f32 -> f16 transpose convert (2D) ----------------
__global__ __launch_bounds__(256) void nm_cvt_f16_t(const float* __restrict__ src,
                                                    _Float16* __restrict__ dst,
                                                    int rows, int cols) {
  int i = blockIdx.x * 256 + threadIdx.x;
  if (i < rows * cols) {
    int r = i / cols, c = i % cols;
    dst[(size_t)c * rows + r] = (_Float16)src[i];
  }
}
// ---------------- Kernel 0c: f32 -> f16 batched 64x64 transpose ----------------
__global__ __launch_bounds__(256) void nm_cvt_f16_t64(const float* __restrict__ src,
                                                      _Float16* __restrict__ dst, int count) {
  int i = blockIdx.x * 256 + threadIdx.x;
  if (i < count) {
    int base = i & ~4095, m = (i >> 6) & 63, n = i & 63;
    dst[base + n * 64 + m] = (_Float16)src[i];
  }
}

// ---------------- Kernel 1: RMSNorm -> f16 s ----------------
__global__ __launch_bounds__(256) void nm_rmsnorm(const float* __restrict__ seq,
                                                  const float* __restrict__ scale,
                                                  _Float16* __restrict__ s16) {
  int row = blockIdx.x;
  const float* src = seq + (size_t)row * DIM;
  __shared__ float red[256];
  int tid = threadIdx.x;
  float v0 = src[tid], v1 = src[tid + 256];
  red[tid] = v0 * v0 + v1 * v1;
  __syncthreads();
  for (int off = 128; off > 0; off >>= 1) {
    if (tid < off) red[tid] += red[tid + off];
    __syncthreads();
  }
  float r = rsqrtf(red[0] / (float)DIM + 1e-6f);
  s16[(size_t)row * DIM + tid]       = (_Float16)(v0 * r * scale[tid]);
  s16[(size_t)row * DIM + tid + 256] = (_Float16)(v1 * r * scale[tid + 256]);
}

// ---------------- Kernel 2: KV GEMM: s(32768x512) @ w_kv(512x1024) ----------------
__global__ __launch_bounds__(256) void nm_kv_gemm(const _Float16* __restrict__ s16,
                                                  const _Float16* __restrict__ wkvT,
                                                  _Float16* __restrict__ k_ws,
                                                  _Float16* __restrict__ v_ws) {
  int rt = blockIdx.x;          // 0..511 : one (b, chunk)
  int ct = blockIdx.y;          // 0..15  : 64-col tile (8 k-heads then 8 v-heads)
  int tid = threadIdx.x, lane = tid & 31, wv = tid >> 5;
  __shared__ __align__(16) _Float16 As[64 * 64];     // row-major A tile
  __shared__ __align__(16) _Float16 BsT[64 * 64];    // column-major B tile
  int t0 = wv * 2, t1 = t0 + 1;
  int m0 = (t0 >> 2) * 16;
  int n00 = (t0 & 3) * 16, n01 = (t1 & 3) * 16;
  v8f acc0 = {}, acc1 = {};
  int row0 = rt * 64;
  for (int kb = 0; kb < 8; ++kb) {
    for (int i = tid; i < 512; i += 256) {   // async global -> LDS tile staging
      int r = i >> 3, p = i & 7;
      async_ld_b128(As + r * 64 + p * 8,
                    s16 + (size_t)(row0 + r) * DIM + kb * 64 + p * 8);
      async_ld_b128(BsT + r * 64 + p * 8,
                    wkvT + (size_t)(ct * 64 + r) * DIM + kb * 64 + p * 8);
    }
    wait_async0();
    __syncthreads();
#pragma unroll
    for (int kk = 0; kk < 64; kk += 32) {
      v16h a = pack_a(As, 64, m0, kk, lane);
      acc0 = wmma16(a, pack_bt(BsT, 64, kk, n00, lane), acc0);
      acc1 = wmma16(a, pack_bt(BsT, 64, kk, n01, lane), acc1);
    }
    __syncthreads();
  }
  int b = rt >> 6, ci = rt & 63;
  int h = (ct < 8) ? ct : ct - 8;
  int bh = b * NH + h;
  _Float16* dst = ((ct < 8) ? k_ws : v_ws) + ((size_t)(ci * BH + bh)) * 4096;
  int nn = n00 + (lane & 15), mb = m0 + ((lane >> 4) << 3);
#pragma unroll
  for (int r = 0; r < 8; ++r)
    dst[(mb + r) * 64 + nn] = (_Float16)fminf(fmaxf(acc0[r], -10.f), 10.f);
  nn = n01 + (lane & 15);
#pragma unroll
  for (int r = 0; r < 8; ++r)
    dst[(mb + r) * 64 + nn] = (_Float16)fminf(fmaxf(acc1[r], -10.f), 10.f);
}

// ---------------- Kernel 3: pooled attention, seq_mean, amom/decay, lr_mean -------------
__global__ __launch_bounds__(256) void nm_pool(const _Float16* __restrict__ s16,
                                               const _Float16* __restrict__ wp1T,
                                               const _Float16* __restrict__ wastepT,
                                               const float* __restrict__ w_pool2,
                                               const float* __restrict__ w_mom,
                                               const float* __restrict__ w_decay,
                                               float* __restrict__ amom_ws,
                                               float* __restrict__ decay_ws,
                                               float* __restrict__ lr_mean) {
  int ci = blockIdx.x, b = blockIdx.y;
  int row0 = b * SEQ + ci * CHK;
  int tid = threadIdx.x, lane = tid & 31, wv = tid >> 5;
  __shared__ __align__(16) _Float16 As[64 * 64];     // s chunk tile (row-major)
  __shared__ __align__(16) _Float16 BsT[64 * 64];    // w_pool1 tile (col-major)
  __shared__ __align__(16) _Float16 CsT[16 * 64];    // w_astep tile (col-major)
  __shared__ float Ps[64 * 64];
  __shared__ float logit[64], attnv[64], sm[DIM], redbuf[2];
  __shared__ float lrred[64 * 16];

  int t0 = wv * 2, t1 = t0 + 1;
  int m0 = (t0 >> 2) * 16;
  int n00 = (t0 & 3) * 16, n01 = (t1 & 3) * 16;
  v8f acc0 = {}, acc1 = {}, acc2 = {};
  for (int kb = 0; kb < 8; ++kb) {
    for (int i = tid; i < 512; i += 256) {
      int r = i >> 3, p = i & 7;
      async_ld_b128(As + r * 64 + p * 8,
                    s16 + (size_t)(row0 + r) * DIM + kb * 64 + p * 8);
      async_ld_b128(BsT + r * 64 + p * 8,
                    wp1T + (size_t)r * DIM + kb * 64 + p * 8);
    }
    if (tid < 128) {
      int r = tid >> 3, p = tid & 7;
      async_ld_b128(CsT + r * 64 + p * 8,
                    wastepT + (size_t)r * DIM + kb * 64 + p * 8);
    }
    wait_async0();
    __syncthreads();
#pragma unroll
    for (int kk = 0; kk < 64; kk += 32) {
      v16h a = pack_a(As, 64, m0, kk, lane);
      acc0 = wmma16(a, pack_bt(BsT, 64, kk, n00, lane), acc0);
      acc1 = wmma16(a, pack_bt(BsT, 64, kk, n01, lane), acc1);
      if (wv < 4) {   // alr logits: 64x16 GEMM against w_astep
        v16h aa = pack_a(As, 64, wv * 16, kk, lane);
        acc2 = wmma16(aa, pack_bt(CsT, 64, kk, 0, lane), acc2);
      }
    }
    __syncthreads();
  }
  {   // silu epilogue (pool) + sigmoid epilogue (alr)
    int nn = n00 + (lane & 15), mb = m0 + ((lane >> 4) << 3);
#pragma unroll
    for (int r = 0; r < 8; ++r) { float h = acc0[r]; Ps[(mb + r) * 64 + nn] = h * sigmoidf_(h); }
    nn = n01 + (lane & 15);
#pragma unroll
    for (int r = 0; r < 8; ++r) { float h = acc1[r]; Ps[(mb + r) * 64 + nn] = h * sigmoidf_(h); }
    if (wv < 4) {
      int hd = lane & 15, mb2 = wv * 16 + ((lane >> 4) << 3);
#pragma unroll
      for (int r = 0; r < 8; ++r)
        lrred[(mb2 + r) * 16 + hd] = sigmoidf_(acc2[r]) * 0.01f;
    }
  }
  __syncthreads();
  if (tid < 64) {
    float a = 0.f;
    for (int nidx = 0; nidx < 64; ++nidx) a += Ps[tid * 64 + nidx] * w_pool2[nidx];
    logit[tid] = a;
  }
  __syncthreads();
  if (tid == 0) { float mx = logit[0]; for (int i = 1; i < 64; ++i) mx = fmaxf(mx, logit[i]); redbuf[0] = mx; }
  __syncthreads();
  if (tid < 64) attnv[tid] = __expf(logit[tid] - redbuf[0]);
  __syncthreads();
  if (tid == 0) { float s = 0.f; for (int i = 0; i < 64; ++i) s += attnv[i]; redbuf[1] = s; }
  __syncthreads();
  if (tid < 64) attnv[tid] /= redbuf[1];
  __syncthreads();
  for (int c = tid; c < DIM; c += 256) {        // seq_mean (attn-weighted)
    float a = 0.f;
    for (int t = 0; t < 64; ++t) a += attnv[t] * (float)s16[(size_t)(row0 + t) * DIM + c];
    sm[c] = a;
  }
  __syncthreads();
  if (tid < 16) {                               // amom / decay
    int h = tid >> 1, which = tid & 1;
    const float* w = which ? w_decay : w_mom;
    float a = 0.f;
    for (int c = 0; c < DIM; ++c) a += sm[c] * w[c * NH + h];
    float sg = sigmoidf_(a);
    int bh = b * NH + h;
    if (which) decay_ws[bh * NCH + ci] = sg; else amom_ws[bh * NCH + ci] = sg;
  }
  if (tid < 16) {                               // lr_mean over tokens
    int h = tid >> 1, d = tid & 1;
    float a = 0.f;
    for (int t = 0; t < 64; ++t) a += lrred[t * 16 + tid];
    int bh = b * NH + h;
    lr_mean[((size_t)ci * BH + bh) * 2 + d] = a / 64.f;
  }
}

// ---------------- Kernel 4: fused TTT gradients (5 WMMA GEMMs per (chunk,bh)) -----------
__global__ __launch_bounds__(256) void nm_ttt_grad(const _Float16* __restrict__ k_ws,
                                                   const _Float16* __restrict__ v_ws,
                                                   const _Float16* __restrict__ w0T,
                                                   const _Float16* __restrict__ w1T,
                                                   const _Float16* __restrict__ w1_16,
                                                   const float* __restrict__ lr_mean,
                                                   float* __restrict__ sp_ws) {
  int ci = blockIdx.x, bh = blockIdx.y;
  int tid = threadIdx.x, lane = tid & 31, wv = tid >> 5;
  // 8 x 8KB = 64KB, with aliasing
  __shared__ __align__(16) _Float16 Ks[4096];    // K row-major; aliased as EsT after GEMM2
  __shared__ __align__(16) _Float16 KsT[4096];   // K col-major (for K^T GEMM)
  __shared__ __align__(16) _Float16 Vs[4096];    // V; aliased as Es after GEMM2
  __shared__ __align__(16) _Float16 W0c[4096];   // W0 col-major; aliased as DHc after GEMM4
  __shared__ __align__(16) _Float16 W1x[4096];   // W1 col-major, restaged row-major for GEMM4
  __shared__ __align__(16) _Float16 Am[4096];    // silu(h) row-major
  __shared__ __align__(16) _Float16 AmT[4096];   // silu(h) col-major
  __shared__ __align__(16) _Float16 Dm[4096];    // silu'(h) row-major
  _Float16* Es  = Vs;
  _Float16* EsT = Ks;
  _Float16* DHc = W0c;
  {   // async stage of K, V, W0(col), W1(col)
    const _Float16* sk = k_ws + ((size_t)ci * BH + bh) * 4096;
    const _Float16* sv = v_ws + ((size_t)ci * BH + bh) * 4096;
    const _Float16* s0 = w0T + (size_t)bh * 4096;
    const _Float16* s1 = w1T + (size_t)bh * 4096;
    for (int i = tid; i < 512; i += 256) {
      int o = i * 8;
      async_ld_b128(Ks + o,  sk + o);
      async_ld_b128(Vs + o,  sv + o);
      async_ld_b128(W0c + o, s0 + o);
      async_ld_b128(W1x + o, s1 + o);
    }
    wait_async0();
  }
  __syncthreads();
  {   // build K^T from LDS
    for (int i = tid; i < 512; i += 256) {
      int r = i >> 3, p = (i & 7) * 8;
      v8h hv = *(const v8h*)(Ks + r * 64 + p);
#pragma unroll
      for (int c = 0; c < 8; ++c) KsT[(p + c) * 64 + r] = hv[c];
    }
  }
  __syncthreads();
  int t0 = wv * 2, t1 = t0 + 1;
  int m0 = (t0 >> 2) * 16;
  int n00 = (t0 & 3) * 16, n01 = (t1 & 3) * 16;
  float lr0 = lr_mean[((size_t)ci * BH + bh) * 2 + 0];
  float lr1 = lr_mean[((size_t)ci * BH + bh) * 2 + 1];
  int nnA = n00 + (lane & 15), nnB = n01 + (lane & 15), mb = m0 + ((lane >> 4) << 3);

  // GEMM1: H = K @ W0 ; Am = silu(H) (both layouts), Dm = silu'(H)
  {
    v8f a0 = {}, a1 = {};
#pragma unroll
    for (int kk = 0; kk < 64; kk += 32) {
      v16h a = pack_a(Ks, 64, m0, kk, lane);
      a0 = wmma16(a, pack_bt(W0c, 64, kk, n00, lane), a0);
      a1 = wmma16(a, pack_bt(W0c, 64, kk, n01, lane), a1);
    }
#pragma unroll
    for (int r = 0; r < 8; ++r) {
      float h = a0[r], sg = sigmoidf_(h);
      _Float16 av = (_Float16)(h * sg);
      Am[(mb + r) * 64 + nnA] = av;
      AmT[nnA * 64 + mb + r]  = av;
      Dm[(mb + r) * 64 + nnA] = (_Float16)(sg * (1.f + h * (1.f - sg)));
      h = a1[r]; sg = sigmoidf_(h);
      av = (_Float16)(h * sg);
      Am[(mb + r) * 64 + nnB] = av;
      AmT[nnB * 64 + mb + r]  = av;
      Dm[(mb + r) * 64 + nnB] = (_Float16)(sg * (1.f + h * (1.f - sg)));
    }
  }
  __syncthreads();
  // GEMM2: P = Am @ W1 ; E = (P - V) * (2/DH)  (both layouts)
  {
    v8f a0 = {}, a1 = {};
#pragma unroll
    for (int kk = 0; kk < 64; kk += 32) {
      v16h a = pack_a(Am, 64, m0, kk, lane);
      a0 = wmma16(a, pack_bt(W1x, 64, kk, n00, lane), a0);
      a1 = wmma16(a, pack_bt(W1x, 64, kk, n01, lane), a1);
    }
    const float c = 2.f / (float)DH;
#pragma unroll
    for (int r = 0; r < 8; ++r) {
      _Float16 ev = (_Float16)((a0[r] - (float)Vs[(mb + r) * 64 + nnA]) * c);
      Es[(mb + r) * 64 + nnA] = ev;
      EsT[nnA * 64 + mb + r]  = ev;
      ev = (_Float16)((a1[r] - (float)Vs[(mb + r) * 64 + nnB]) * c);
      Es[(mb + r) * 64 + nnB] = ev;
      EsT[nnB * 64 + mb + r]  = ev;
    }
  }
  __syncthreads();
  // restage W1 row-major (for W1^T operand of GEMM4), async
  {
    const _Float16* s1r = w1_16 + (size_t)bh * 4096;
    for (int i = tid; i < 512; i += 256) async_ld_b128(W1x + i * 8, s1r + i * 8);
    wait_async0();
  }
  __syncthreads();
  // GEMM3: G1 = Am^T @ E  -> sp1 = clean(-G1 * lr1)
  {
    v8f a0 = {}, a1 = {};
#pragma unroll
    for (int kk = 0; kk < 64; kk += 32) {
      v16h a = pack_a(AmT, 64, m0, kk, lane);
      a0 = wmma16(a, pack_bt(EsT, 64, kk, n00, lane), a0);
      a1 = wmma16(a, pack_bt(EsT, 64, kk, n01, lane), a1);
    }
    float* dst = sp_ws + ((size_t)(BH + bh) * NCH + ci) * 4096;
#pragma unroll
    for (int r = 0; r < 8; ++r) {
      dst[(mb + r) * 64 + nnA] = cleanf_(-a0[r] * lr1);
      dst[(mb + r) * 64 + nnB] = cleanf_(-a1[r] * lr1);
    }
  }
  // GEMM4: DA = E @ W1^T ; dh = DA * Dm -> DHc (col-major)
  {
    v8f a0 = {}, a1 = {};
#pragma unroll
    for (int kk = 0; kk < 64; kk += 32) {
      v16h a = pack_a(Es, 64, m0, kk, lane);
      a0 = wmma16(a, pack_bt(W1x, 64, kk, n00, lane), a0);   // row-major W1 => B = W1^T
      a1 = wmma16(a, pack_bt(W1x, 64, kk, n01, lane), a1);
    }
#pragma unroll
    for (int r = 0; r < 8; ++r) {
      DHc[nnA * 64 + mb + r] = (_Float16)(a0[r] * (float)Dm[(mb + r) * 64 + nnA]);
      DHc[nnB * 64 + mb + r] = (_Float16)(a1[r] * (float)Dm[(mb + r) * 64 + nnB]);
    }
  }
  __syncthreads();
  // GEMM5: G0 = K^T @ dh -> sp0 = clean(-G0 * lr0)
  {
    v8f a0 = {}, a1 = {};
#pragma unroll
    for (int kk = 0; kk < 64; kk += 32) {
      v16h a = pack_a(KsT, 64, m0, kk, lane);
      a0 = wmma16(a, pack_bt(DHc, 64, kk, n00, lane), a0);
      a1 = wmma16(a, pack_bt(DHc, 64, kk, n01, lane), a1);
    }
    float* dst = sp_ws + ((size_t)bh * NCH + ci) * 4096;
#pragma unroll
    for (int r = 0; r < 8; ++r) {
      dst[(mb + r) * 64 + nnA] = cleanf_(-a0[r] * lr0);
      dst[(mb + r) * 64 + nnB] = cleanf_(-a1[r] * lr0);
    }
  }
}

// ---------------- Kernel 5: double linear-recurrence scan over chunks -------------------
__global__ __launch_bounds__(256) void nm_scan(const float* __restrict__ sp,
                                               const float* __restrict__ amom,
                                               const float* __restrict__ decay,
                                               float* __restrict__ out) {
  int gb = blockIdx.x;              // 0..127 : g*64 + bh
  int g = gb >> 6, bh = gb & 63;
  int elem = blockIdx.y * 256 + threadIdx.x;   // 0..4095
  const float* spb = sp + ((size_t)(g * BH + bh) * NCH) * 4096 + elem;
  float mom = 0.f, upd = 0.f;
  for (int t = 0; t < NCH; ++t) {
    float s  = spb[(size_t)t * 4096];
    float am = amom[bh * NCH + t];
    float dc = decay[bh * NCH + t];
    mom = am * mom + s;
    upd = (1.f - dc) * upd + mom;
    out[(((size_t)g       * BH + bh) * NCH + t) * 4096 + elem] = upd;  // slots 0,1
    out[(((size_t)(g + 2) * BH + bh) * NCH + t) * 4096 + elem] = mom;  // slots 2,3
  }
}

// ---------------- host launcher ----------------
extern "C" void kernel_launch(void* const* d_in, const int* in_sizes, int n_in,
                              void* d_out, int out_size, void* d_ws, size_t ws_size,
                              hipStream_t stream) {
  const float* seq        = (const float*)d_in[0];
  const float* norm_scale = (const float*)d_in[1];
  const float* w_kv       = (const float*)d_in[2];
  const float* w_astep    = (const float*)d_in[3];
  const float* w_mom      = (const float*)d_in[4];
  const float* w_decay    = (const float*)d_in[5];
  const float* w_pool1    = (const float*)d_in[6];
  const float* w_pool2    = (const float*)d_in[7];
  const float* past_w0    = (const float*)d_in[8];
  const float* past_w1    = (const float*)d_in[9];
  float* out = (float*)d_out;

  char* ws = (char*)d_ws;
  _Float16* s16      = (_Float16*)(ws);                        // 32 MB
  _Float16* wkvT     = (_Float16*)(ws + 33554432);             // 1 MB   [1024][512]
  _Float16* wp1T     = (_Float16*)(ws + 34603008);             // 64 KB  [64][512]
  _Float16* wastepT  = (_Float16*)(ws + 34668544);             // 16 KB  [16][512]
  _Float16* w1_16    = (_Float16*)(ws + 34684928);             // 512 KB row-major
  _Float16* w0T      = (_Float16*)(ws + 35209216);             // 512 KB col-major
  _Float16* w1T      = (_Float16*)(ws + 35733504);             // 512 KB col-major
  _Float16* k_ws     = (_Float16*)(ws + 36257792);             // 32 MB
  _Float16* v_ws     = (_Float16*)(ws + 69812224);             // 32 MB
  float*    lr_mean  = (float*)   (ws + 103366656);            // 32 KB
  float*    amom_ws  = (float*)   (ws + 103399424);            // 16 KB
  float*    decay_ws = (float*)   (ws + 103415808);            // 16 KB
  float*    sp_ws    = (float*)   (ws + 103432192);            // 128 MB

  // weight converts (row-major + transposed-for-B-operand layouts)
  nm_cvt_f16_t<<<(DIM * 1024 + 255) / 256, 256, 0, stream>>>(w_kv, wkvT, DIM, 1024);
  nm_cvt_f16_t<<<(DIM * 64 + 255) / 256, 256, 0, stream>>>(w_pool1, wp1T, DIM, 64);
  nm_cvt_f16_t<<<(DIM * 16 + 255) / 256, 256, 0, stream>>>(w_astep, wastepT, DIM, 16);
  nm_cvt_f16<<<(BH * 4096 + 255) / 256, 256, 0, stream>>>(past_w1, w1_16, BH * 4096);
  nm_cvt_f16_t64<<<(BH * 4096 + 255) / 256, 256, 0, stream>>>(past_w0, w0T, BH * 4096);
  nm_cvt_f16_t64<<<(BH * 4096 + 255) / 256, 256, 0, stream>>>(past_w1, w1T, BH * 4096);

  // RMSNorm
  nm_rmsnorm<<<ROWS, 256, 0, stream>>>(seq, norm_scale, s16);

  // KV GEMM (k/v chunked + clipped)
  nm_kv_gemm<<<dim3(ROWS / 64, 16), 256, 0, stream>>>(s16, wkvT, k_ws, v_ws);

  // pooled attention + stats + alr (WMMA)
  nm_pool<<<dim3(NCH, BSZ), 256, 0, stream>>>(s16, wp1T, wastepT, w_pool2, w_mom,
                                              w_decay, amom_ws, decay_ws, lr_mean);

  // fused TTT gradients
  nm_ttt_grad<<<dim3(NCH, BH), 256, 0, stream>>>(k_ws, v_ws, w0T, w1T, w1_16,
                                                 lr_mean, sp_ws);

  // scans -> output (4, B, H, n, DH, DH)
  nm_scan<<<dim3(2 * BH, 16), 256, 0, stream>>>(sp_ws, amom_ws, decay_ws, out);
}
